// LogFlashAttentionNd_35459249996585
// MI455X (gfx1250) — matrix operand order repaired and checked by reading, not compiled, not run on hardware
//
#include <hip/hip_runtime.h>
#include <hip/hip_bf16.h>

typedef __attribute__((ext_vector_type(16))) __bf16 v16bf;
typedef __attribute__((ext_vector_type(8)))  float  v8f;

#define BN_INV 0.99999500003749969f  // 1/sqrt(1+1e-5)

__device__ __forceinline__ void pack8(v16bf& d, int base, float4 u, float4 v) {
  d[base + 0] = (__bf16)u.x; d[base + 1] = (__bf16)u.y;
  d[base + 2] = (__bf16)u.z; d[base + 3] = (__bf16)u.w;
  d[base + 4] = (__bf16)v.x; d[base + 5] = (__bf16)v.y;
  d[base + 6] = (__bf16)v.z; d[base + 7] = (__bf16)v.w;
}

// ---------------------------------------------------------------------------
// WMMA GEMM + BatchNorm:  dst = BN( W(O x 256) * X[b](256 x 4096) )
// One wave computes a 16(o) x 64(p) strip: A (weights) loaded once per K-step
// as float4 pairs, 4 WMMAs issued against it. Next K-slice of X prefetched.
// mode 0: dst layout [b*8+head][p][t][d]   (qk, O==512, t=q/k)
// mode 1: dst layout [b][o][p]             (v0 / proj / final output)
// ---------------------------------------------------------------------------
__global__ void __launch_bounds__(128) k_gemm_bn(
    const float* __restrict__ X,   // [4][256][4096]
    const float* __restrict__ W,   // [O][256]
    const float* __restrict__ G,
    const float* __restrict__ Bi,
    float* __restrict__ dst, int O, int mode)
{
  int lane = threadIdx.x & 31, wv = threadIdx.x >> 5;
  int lr = lane & 15, hi = lane >> 4;
  int p0 = blockIdx.x * 256 + wv * 64;   // wave's 64-position strip
  int ot = blockIdx.y * 16;
  int b  = blockIdx.z;
  const float* Xb = X + (size_t)b * 256 * 4096;

  v8f z8 = {0.f,0.f,0.f,0.f,0.f,0.f,0.f,0.f};
  v8f acc[4] = {z8, z8, z8, z8};
  int arow = ot + lr;
  const float* wrow = W + (size_t)arow * 256;

  for (int kk = 0; kk < 256; kk += 32) {
    // A operand (16x32 bf16): two contiguous 8-float runs per lane -> b128 loads
    v16bf a;
    {
      const float4* w0 = (const float4*)(wrow + kk + hi * 8);
      const float4* w1 = (const float4*)(wrow + kk + 16 + hi * 8);
      pack8(a, 0, w0[0], w0[1]);
      pack8(a, 8, w1[0], w1[1]);
    }
    // speculative prefetch of next K-slice of X (global_prefetch_b8)
    int nk = (kk + 32 < 256) ? kk + 32 : kk;
    __builtin_prefetch(&Xb[(size_t)(nk + hi * 16) * 4096 + p0 + lr], 0, 1);

#pragma unroll
    for (int j = 0; j < 4; ++j) {
      int col = p0 + j * 16 + lr;
      v16bf bm;
#pragma unroll
      for (int e = 0; e < 16; ++e) {
        int cB = kk + (e >> 3) * 16 + hi * 8 + (e & 7);  // B layout K index
        bm[e] = (__bf16)Xb[(size_t)cB * 4096 + col];
      }
      acc[j] = __builtin_amdgcn_wmma_f32_16x16x32_bf16(false, a, false, bm,
                                                       (short)0, acc[j], false, false);
    }
  }

  // BN epilogue: hoist per-channel scale/bias (channel = ot + r + 8*hi)
  float sc8[8], bi8[8];
#pragma unroll
  for (int r = 0; r < 8; ++r) {
    int o = ot + r + 8 * hi;
    sc8[r] = G[o] * BN_INV;
    bi8[r] = Bi[o];
  }
#pragma unroll
  for (int j = 0; j < 4; ++j) {
    int col = p0 + j * 16 + lr;
#pragma unroll
    for (int r = 0; r < 8; ++r) {
      int o = ot + r + 8 * hi;
      float y = acc[j][r] * sc8[r] + bi8[r];
      if (mode == 0) {
        int t = o >> 8, rem = o & 255, head = rem >> 5, d = rem & 31;
        dst[(((size_t)(b * 8 + head) * 4096) + col) * 64 + t * 32 + d] = y;
      } else {
        dst[((size_t)(b * O + o)) * 4096 + col] = y;
      }
    }
  }
}

// ---------------------------------------------------------------------------
// One log-attention axis. Each wave handles one (batch*head, group) problem:
//   S = Q(16x32) K^T  -> mask cols >= bsz -> row softmax -> O = P * V(16x32)
// qk:   [b*8+head][4096][2][32]  (only p<4096 valid; p>=4096 treated as 0)
// vsrc: first axis -> v0 channel-major [b][256][4096]; else [bh][L][32]
// vdst: [bh][L][32]  (in-place across axes; groups are disjoint)
// ---------------------------------------------------------------------------
__global__ void __launch_bounds__(128) k_attn_axis(
    const float* __restrict__ qk,
    const float* __restrict__ vsrc,
    float* __restrict__ vdst,
    int bsz, int L, int stride, int first, int nprob)
{
  __shared__ float sm[4][256];
  int lane = threadIdx.x & 31, wv = threadIdx.x >> 5;
  int lr = lane & 15, hi = lane >> 4;
  int pid = blockIdx.x * 4 + wv;
  if (pid >= nprob) pid = nprob - 1;          // benign duplicate (same value stores)

  int gcount = L / bsz;
  int bh = pid / gcount;
  int g  = pid % gcount;
  int base_p = (g / stride) * (stride * bsz) + (g % stride);
  const float scale = 0.17677669529663687f;   // 1/sqrt(32)

  // --- load Q as A-operand, K^T as B-operand (rows/cols >= bsz -> 0) ---
  int p_row = base_p + lr * stride;
  bool vrow = (lr < bsz) && (p_row < 4096);
  const float* qkrow = qk + (((size_t)bh * 4096) + (size_t)(vrow ? p_row : 0)) * 64;
  v16bf qa, kb;
  if (vrow) {
    const float4* q0 = (const float4*)(qkrow + hi * 8);        // d = hi*8 + 0..7
    const float4* q1 = (const float4*)(qkrow + 16 + hi * 8);   // d = 16+hi*8 + 0..7
    pack8(qa, 0, q0[0], q0[1]);
    pack8(qa, 8, q1[0], q1[1]);
    const float4* k0 = (const float4*)(qkrow + 32 + hi * 8);       // d = hi*8 + 0..7
    const float4* k1 = (const float4*)(qkrow + 48 + hi * 8);       // d = 16+hi*8 + 0..7
    pack8(kb, 0, k0[0], k0[1]);
    pack8(kb, 8, k1[0], k1[1]);
  } else {
#pragma unroll
    for (int e = 0; e < 16; ++e) { qa[e] = (__bf16)0.0f; kb[e] = (__bf16)0.0f; }
  }

  v8f zero8 = {0.f,0.f,0.f,0.f,0.f,0.f,0.f,0.f};
  v8f sc = __builtin_amdgcn_wmma_f32_16x16x32_bf16(false, qa, false, kb,
                                                   (short)0, zero8, false, false);

  // --- row softmax (row spread over 16 lanes of each half-wave) ---
  float prob[8];
#pragma unroll
  for (int r = 0; r < 8; ++r) {
    float s = (lr < bsz) ? sc[r] * scale : -3.0e38f;   // mask padded key cols
    float m = s;
    m = fmaxf(m, __shfl_xor(m, 1));
    m = fmaxf(m, __shfl_xor(m, 2));
    m = fmaxf(m, __shfl_xor(m, 4));
    m = fmaxf(m, __shfl_xor(m, 8));
    float ev = __expf(s - m);
    float su = ev;
    su += __shfl_xor(su, 1);
    su += __shfl_xor(su, 2);
    su += __shfl_xor(su, 4);
    su += __shfl_xor(su, 8);
    prob[r] = ev / su;
  }

  // --- transpose P (C-layout -> row-major LDS -> A-layout) ---
#pragma unroll
  for (int r = 0; r < 8; ++r) sm[wv][(r + 8 * hi) * 16 + lr] = prob[r];
  __syncthreads();
  v16bf pa;
#pragma unroll
  for (int e = 0; e < 16; ++e) {
    int K = ((e >> 3) * 16) + hi * 8 + (e & 7);        // K>=16 is zero padding
    pa[e] = (K < 16) ? (__bf16)sm[wv][lr * 16 + K] : (__bf16)0.0f;
  }

  // --- O = P * V over two 16-wide d-tiles ---
  int b0 = bh >> 3, head = bh & 7;
#pragma unroll
  for (int dt = 0; dt < 2; ++dt) {
    v16bf vb;
#pragma unroll
    for (int e = 0; e < 16; ++e) {
      int t = (e >> 3) * 16 + hi * 8 + (e & 7);        // B layout K index
      float val = 0.f;
      if (t < bsz) {
        int p_v = base_p + t * stride;
        if (first) {
          if (p_v < 4096)
            val = vsrc[((size_t)(b0 * 256 + head * 32 + dt * 16 + lr)) * 4096 + p_v];
        } else {
          val = vsrc[(((size_t)bh) * L + p_v) * 32 + dt * 16 + lr];
        }
      }
      vb[e] = (__bf16)val;
    }
    v8f o = __builtin_amdgcn_wmma_f32_16x16x32_bf16(false, pa, false, vb,
                                                    (short)0, zero8, false, false);
#pragma unroll
    for (int r = 0; r < 8; ++r) {
      int s = r + 8 * hi;
      if (s < bsz) {
        int p_o = base_p + s * stride;
        vdst[(((size_t)bh) * L + p_o) * 32 + dt * 16 + lr] = o[r];
      }
    }
  }
}

// ---------------------------------------------------------------------------
// y[b][c][p] = (init ? x : y) + v2 + BN(dwconv3x3(v0))
// ---------------------------------------------------------------------------
__global__ void __launch_bounds__(256) k_combine(
    const float* __restrict__ x,
    const float* __restrict__ vws,   // [b*8+head][L][32]
    const float* __restrict__ v0c,   // [b][256][4096]
    const float* __restrict__ pew,   // [256][9]
    const float* __restrict__ peg,
    const float* __restrict__ peb,
    float* __restrict__ y, int L, int init)
{
  int idx = blockIdx.x * blockDim.x + threadIdx.x;
  int p = idx & 4095;
  int c = (idx >> 12) & 255;
  int b = idx >> 20;
  int px = p & 63, py = p >> 6;
  int head = c >> 5, d = c & 31;

  float v2 = vws[(((size_t)(b * 8 + head)) * L + p) * 32 + d];

  const float* vc = v0c + ((size_t)(b * 256 + c)) * 4096;
  float acc = 0.f;
#pragma unroll
  for (int dy = -1; dy <= 1; ++dy)
#pragma unroll
    for (int dx = -1; dx <= 1; ++dx) {
      int yy = py + dy, xx = px + dx;
      if (yy >= 0 && yy < 64 && xx >= 0 && xx < 64)
        acc += vc[yy * 64 + xx] * pew[c * 9 + (dy + 1) * 3 + (dx + 1)];
    }
  float pe = acc * (peg[c] * BN_INV) + peb[c];
  float base = init ? x[idx] : y[idx];
  y[idx] = base + v2 + pe;
}

// ---------------------------------------------------------------------------
extern "C" void kernel_launch(void* const* d_in, const int* in_sizes, int n_in,
                              void* d_out, int out_size, void* d_ws, size_t ws_size,
                              hipStream_t stream)
{
  const float* x      = (const float*)d_in[0];
  const float* proj_w = (const float*)d_in[1];
  const float* proj_g = (const float*)d_in[2];
  const float* proj_b = (const float*)d_in[3];
  const float* qk_w[2] = {(const float*)d_in[4],  (const float*)d_in[13]};
  const float* qk_g[2] = {(const float*)d_in[5],  (const float*)d_in[14]};
  const float* qk_b[2] = {(const float*)d_in[6],  (const float*)d_in[15]};
  const float* v_w[2]  = {(const float*)d_in[7],  (const float*)d_in[16]};
  const float* v_g[2]  = {(const float*)d_in[8],  (const float*)d_in[17]};
  const float* v_b[2]  = {(const float*)d_in[9],  (const float*)d_in[18]};
  const float* pe_w[2] = {(const float*)d_in[10], (const float*)d_in[19]};
  const float* pe_g[2] = {(const float*)d_in[11], (const float*)d_in[20]};
  const float* pe_b[2] = {(const float*)d_in[12], (const float*)d_in[21]};

  float* ws    = (float*)d_ws;
  float* qk_ws = ws;                    // 4*8*4096*64   =  8,388,608 f
  float* v0c   = qk_ws + 8388608;       // 4*256*4096    =  4,194,304 f
  float* vws   = v0c   + 4194304;       // 4*8*15625*32  = 16,000,000 f (max L)
  float* y_ws  = vws   + 16000000;      // 4*256*4096    =  4,194,304 f

  const int bases[2] = {16, 5};
  const int ns[2]    = {3, 6};
  const int Ls[2]    = {4096, 15625};
  dim3 blk(128);

  for (int br = 0; br < 2; ++br) {
    // qk = BN(conv1x1) -> attention layout; v0 = BN(conv1x1) -> channel-major
    k_gemm_bn<<<dim3(16, 32, 4), blk, 0, stream>>>(x, qk_w[br], qk_g[br], qk_b[br],
                                                   qk_ws, 512, 0);
    k_gemm_bn<<<dim3(16, 16, 4), blk, 0, stream>>>(x, v_w[br], v_g[br], v_b[br],
                                                   v0c, 256, 1);
    int bsz = bases[br], n = ns[br], L = Ls[br];
    int stride = L / bsz;               // bsz^(n-1), outermost digit first
    for (int ax = 0; ax < n; ++ax) {
      int nprob = 32 * (L / bsz);       // (4 batches * 8 heads) * groups
      k_attn_axis<<<dim3((nprob + 3) / 4), blk, 0, stream>>>(
          qk_ws, ax == 0 ? v0c : vws, vws, bsz, L, stride, ax == 0 ? 1 : 0, nprob);
      stride /= bsz;
    }
    k_combine<<<dim3(4194304 / 256), dim3(256), 0, stream>>>(
        x, vws, v0c, pe_w[br], pe_g[br], pe_b[br], y_ws, L, br == 0 ? 1 : 0);
  }

  // final projection: out = BN(conv1x1(y, proj_w)) straight into d_out
  k_gemm_bn<<<dim3(16, 16, 4), blk, 0, stream>>>(y_ws, proj_w, proj_g, proj_b,
                                                 (float*)d_out, 256, 1);
}
